// GlmMoeDsaAttention_33371895890038
// MI455X (gfx1250) — matrix operand order repaired and checked by reading, not compiled, not run on hardware
//
#include <hip/hip_runtime.h>
#include <math.h>

// ---------------------------------------------------------------------------
// GLM MoE DSA attention for gfx1250 (MI455X), bf16 WMMA pipeline.
// All matrix math lowers to v_wmma_f32_16x16x32_bf16 (f32 accumulate).
// GEMMs are register-blocked M16xN(16*NT) per wave with NT a compile-time
// template parameter -> straight-line unrolled WMMA bodies (no phi copies).
// ---------------------------------------------------------------------------

typedef __attribute__((ext_vector_type(16))) __bf16 v16bf_t;
typedef __attribute__((ext_vector_type(8)))  float  v8f_t;

namespace dsa {
constexpr int T    = 1024;
constexpr int HID  = 2048;
constexpr int H    = 16;
constexpr int LQ   = 1536;
constexpr int LKV  = 512;
constexpr int DR   = 64;
constexpr int DN   = 128;
constexpr int DQ   = 192;
constexpr int DV   = 128;
constexpr int HI   = 16;
constexpr int DI   = 128;
constexpr int TOPK = 256;
constexpr float NEGV   = -1e30f;
constexpr float EPSV   = 1e-6f;
constexpr float WSCALE = 0.022097086912079608f;   // DI^-0.5 * HI^-0.5
constexpr float QSCALE = 0.072168783648703332f;   // DQ^-0.5  (DQ = 192)
}

// ---------------- WMMA fragment helpers (wave32) ----------------------------
// A matrix (16x32 bf16): lanes 0-15 row M=lane, K = 0..15 sequential;
//                        lanes 16-31 row M=lane-16, K = 16..31.
__device__ __forceinline__ v16bf_t load_a_rm(const float* base, long ld,
                                             int lane, int k0) {
  const float* r = base + (long)(lane & 15) * ld;
  int kb = k0 + ((lane >> 4) ? 16 : 0);
  v16bf_t f;
#pragma unroll
  for (int j = 0; j < 8; ++j) {
    int kk = kb + 2 * j;
    f[2 * j]     = (__bf16)r[kk];
    f[2 * j + 1] = (__bf16)r[kk + 1];
  }
  return f;
}

// B matrix (32x16 bf16): lane = column N; lanes 0-15 hold K=0..15 sequential,
// lanes 16-31 hold K=16..31.  Source stored row-major [N, K] (i.e. B^T rows).
__device__ __forceinline__ v16bf_t load_b_rm(const float* base, long ld,
                                             int lane, int k0) {
  const float* r = base + (long)(lane & 15) * ld;
  int kb = k0 + ((lane >> 4) ? 16 : 0);
  v16bf_t f;
#pragma unroll
  for (int j = 0; j < 8; ++j) {
    int kk = kb + 2 * j;
    f[2 * j]     = (__bf16)r[kk];
    f[2 * j + 1] = (__bf16)r[kk + 1];
  }
  return f;
}

__device__ __forceinline__ v8f_t wmma_bf16(v16bf_t a, v16bf_t b, v8f_t c) {
  return __builtin_amdgcn_wmma_f32_16x16x32_bf16(false, a, false, b,
                                                 (short)0, c, false, false);
}

// ---------------- Generic GEMM: C[M,N] = A[M,K] @ B^T ----------------------
// One wave per 16x(16*NT) output block; A fragment reused NT times.
// B stored row-major [N,K].  blockIdx.z batches via element strides.
template <int NT>
__global__ void dsa_gemm_bf16(const float* __restrict__ A, long lda, long sAz,
                              const float* __restrict__ B, long ldb, long sBz,
                              float* __restrict__ C, long ldc, long sCz,
                              int K) {
  int lane = threadIdx.x;
  long n0 = (long)blockIdx.x * (16 * NT);
  long m0 = (long)blockIdx.y * 16;
  A += (long)blockIdx.z * sAz;
  B += (long)blockIdx.z * sBz;
  C += (long)blockIdx.z * sCz;

  const float* Arow = A + m0 * lda;
  v8f_t acc[NT];
#pragma unroll
  for (int st = 0; st < NT; ++st) acc[st] = (v8f_t){};

  for (int k0 = 0; k0 < K; k0 += 32) {
    if (k0 + 32 < K)  // warm next A chunk (global_prefetch_b8)
      __builtin_prefetch(Arow + (long)(lane & 15) * lda + k0 + 32, 0, 1);
    v16bf_t a = load_a_rm(Arow, lda, lane, k0);
#pragma unroll
    for (int st = 0; st < NT; ++st) {
      v16bf_t b = load_b_rm(B + (n0 + 16 * st) * ldb, ldb, lane, k0);
      acc[st] = wmma_bf16(a, b, acc[st]);
    }
  }
  int mOff = (lane >> 4) * 8;
  int n = lane & 15;
#pragma unroll
  for (int st = 0; st < NT; ++st)
#pragma unroll
    for (int i = 0; i < 8; ++i)
      C[(m0 + i + mOff) * ldc + n0 + 16 * st + n] = acc[st][i];
}

// ---------------- one-time transpose: w_kn[h] (DN x LKV) -> (LKV x DN) -----
// wkt[h][n][k] = w_kvb[(h*(DN+DV) + k)*LKV + n]
__global__ void dsa_transpose_wkn(const float* __restrict__ w_kvb,
                                  float* __restrict__ wkt) {
  using namespace dsa;
  int h = blockIdx.y;
  int i = blockIdx.x * 256 + threadIdx.x;   // over LKV*DN per head
  int n = i >> 7;           // 0..511
  int k = i & 127;          // 0..127
  wkt[((long)h * LKV + n) * DN + k] =
      w_kvb[((long)(h * (DN + DV)) + k) * LKV + n];
}

// ---------------- RMSNorm / LayerNorm / rope / zero ------------------------
__global__ void dsa_rms(float* __restrict__ X, long rowStride, int C,
                        const float* __restrict__ g) {
  int t = blockIdx.x;
  float* row = X + (long)t * rowStride;
  __shared__ float red[256];
  float s = 0.f;
  for (int c = threadIdx.x; c < C; c += blockDim.x) { float v = row[c]; s += v * v; }
  red[threadIdx.x] = s; __syncthreads();
  for (int st = blockDim.x / 2; st > 0; st >>= 1) {
    if ((int)threadIdx.x < st) red[threadIdx.x] += red[threadIdx.x + st];
    __syncthreads();
  }
  float scale = rsqrtf(red[0] / (float)C + dsa::EPSV);
  for (int c = threadIdx.x; c < C; c += blockDim.x) row[c] = row[c] * scale * g[c];
}

__global__ void dsa_ln(float* __restrict__ X, int C,
                       const float* __restrict__ w, const float* __restrict__ b) {
  int t = blockIdx.x;
  float* row = X + (long)t * C;
  __shared__ float red[128];
  float s = 0.f;
  for (int c = threadIdx.x; c < C; c += blockDim.x) s += row[c];
  red[threadIdx.x] = s; __syncthreads();
  for (int st = blockDim.x / 2; st > 0; st >>= 1) {
    if ((int)threadIdx.x < st) red[threadIdx.x] += red[threadIdx.x + st];
    __syncthreads();
  }
  float mean = red[0] / (float)C; __syncthreads();
  float v = 0.f;
  for (int c = threadIdx.x; c < C; c += blockDim.x) { float d = row[c] - mean; v += d * d; }
  red[threadIdx.x] = v; __syncthreads();
  for (int st = blockDim.x / 2; st > 0; st >>= 1) {
    if ((int)threadIdx.x < st) red[threadIdx.x] += red[threadIdx.x + st];
    __syncthreads();
  }
  float inv = rsqrtf(red[0] / (float)C + dsa::EPSV);
  for (int c = threadIdx.x; c < C; c += blockDim.x)
    row[c] = (row[c] - mean) * inv * w[c] + b[c];
}

// rope_interleave in place: out[i]=x[2i]*c[i]-x[2i+1]*s[i];
//                           out[32+i]=x[2i+1]*c[32+i]+x[2i]*s[32+i]
__global__ void dsa_rope(float* __restrict__ X, long rowStride, long headStride,
                         const float* __restrict__ cosp,
                         const float* __restrict__ sinp) {
  int t = blockIdx.x, h = blockIdx.y, i = threadIdx.x;  // 32 threads
  float* p = X + (long)t * rowStride + (long)h * headStride;
  float x1 = p[2 * i], x2 = p[2 * i + 1];
  __syncthreads();
  float c0 = cosp[t * dsa::DR + i],       s0 = sinp[t * dsa::DR + i];
  float c1 = cosp[t * dsa::DR + 32 + i],  s1 = sinp[t * dsa::DR + 32 + i];
  p[i]      = x1 * c0 - x2 * s0;
  p[32 + i] = x2 * c1 + x1 * s1;
}

__global__ void dsa_zero(float* p, int n) {
  int i = blockIdx.x * blockDim.x + threadIdx.x;
  if (i < n) p[i] = 0.f;
}

// ---------------- Fused indexer score (WMMA + relu*w epilogue) -------------
// grid = (T/64 key-blocks, T tokens), 1 wave.  Per block: preload the 4 A
// fragments (16 heads x K=128) once, reuse across 4 key subtiles.
__global__ void dsa_indexer(const float* __restrict__ qi,
                            const float* __restrict__ ki,
                            const float* __restrict__ wproj,
                            const int* __restrict__ ks,
                            const int* __restrict__ ke,
                            float* __restrict__ score) {
  using namespace dsa;
  int lane = threadIdx.x;
  int sbase = blockIdx.x * 64;
  int t = blockIdx.y;
  const float* Abase = qi + (long)t * (HI * DI);   // 16 head rows, ld=DI
  v16bf_t afr[4];
#pragma unroll
  for (int kq = 0; kq < 4; ++kq) afr[kq] = load_a_rm(Abase, DI, lane, kq * 32);

  __shared__ float part[4][32];
  int mOff = (lane >> 4) * 8;
#pragma unroll
  for (int st = 0; st < 4; ++st) {
    int s0 = sbase + 16 * st;
    v8f_t acc = {};
#pragma unroll
    for (int kq = 0; kq < 4; ++kq) {
      v16bf_t b = load_b_rm(ki + (long)s0 * DI, DI, lane, kq * 32);
      acc = wmma_bf16(afr[kq], b, acc);
    }
    float partial = 0.f;
#pragma unroll
    for (int i = 0; i < 8; ++i) {
      float v = acc[i];
      v = v > 0.f ? v : 0.f;                       // relu(logits)
      partial += v * wproj[t * HI + i + mOff];     // * w[t, head]
    }
    part[st][lane] = partial;
  }
  __syncthreads();
  if (lane < 16) {
    int kst = ks[t], ket = ke[t];
#pragma unroll
    for (int st = 0; st < 4; ++st) {
      int s = sbase + 16 * st + lane;
      float val = (part[st][lane] + part[st][lane + 16]) * WSCALE;
      bool valid = (s >= kst) && (s <= ket);
      score[(long)t * T + s] = valid ? val : NEGV;
    }
  }
}

// ---------------- Top-k (iterative argmax, 256 rounds) ---------------------
__global__ void dsa_topk(const float* __restrict__ score, int* __restrict__ idx) {
  using namespace dsa;
  int t = blockIdx.x, tid = threadIdx.x;           // 256 threads
  __shared__ float sv[T];
  __shared__ float rv[256];
  __shared__ int   ra[256];
  for (int c = tid; c < T; c += 256) sv[c] = score[(long)t * T + c];
  __syncthreads();
  for (int r = 0; r < TOPK; ++r) {
    float best = -3.0e38f; int barg = 0;
    for (int c = tid; c < T; c += 256)
      if (sv[c] > best) { best = sv[c]; barg = c; }
    rv[tid] = best; ra[tid] = barg; __syncthreads();
    for (int s = 128; s > 0; s >>= 1) {
      if (tid < s && rv[tid + s] > rv[tid]) { rv[tid] = rv[tid + s]; ra[tid] = ra[tid + s]; }
      __syncthreads();
    }
    if (tid == 0) {
      idx[(long)t * TOPK + r] = (rv[0] <= NEGV * 0.5f) ? T : ra[0];
      sv[ra[0]] = -3.0e38f;
    }
    __syncthreads();
  }
}

// ---------------- Sparse attention core (1 wave / token) -------------------
// Stage1: logits[16,256] = sq(16x576) @ gathered-kv^T, scale + mask.
// Softmax per head (lane-shuffle).  Stage2: out[16,512] = p @ gathered-kv.
__global__ void dsa_attention(const float* __restrict__ qabs,
                              const float* __restrict__ qfull,
                              const float* __restrict__ skv,
                              const int* __restrict__ indices,
                              float* __restrict__ aout) {
  using namespace dsa;
  int t = blockIdx.x, lane = threadIdx.x;
  __shared__ int    sidx[TOPK];
  __shared__ float  sl[H][TOPK];
  __shared__ __bf16 sp[H][TOPK];
  for (int j = lane; j < TOPK; j += 32) sidx[j] = indices[(long)t * TOPK + j];
  __syncthreads();

  constexpr int KVW = LKV + DR;  // 576
  const float* Aabs = qabs + (long)t * H * LKV;                 // ld = LKV
  const float* Arop = qfull + (long)t * (H * DQ) + DN - LKV;    // ld = DQ

  // ---- stage 1 ----
  for (int j0 = 0; j0 < TOPK; j0 += 16) {
    v8f_t acc = {};
    int col = j0 + (lane & 15);
    const float* krow = skv + (long)sidx[col] * KVW;
    for (int k0 = 0; k0 < LKV; k0 += 32) {           // absorbed-nope part
      v16bf_t a = load_a_rm(Aabs, LKV, lane, k0);
      v16bf_t b;
      {
        int kb = k0 + ((lane >> 4) ? 16 : 0);
#pragma unroll
        for (int j = 0; j < 8; ++j) {
          int kk = kb + 2 * j;
          b[2 * j]     = (__bf16)krow[kk];
          b[2 * j + 1] = (__bf16)krow[kk + 1];
        }
      }
      acc = wmma_bf16(a, b, acc);
    }
#pragma unroll
    for (int kq = 0; kq < 2; ++kq) {                 // rope tail (k 512..575)
      int k0 = LKV + kq * 32;
      v16bf_t a = load_a_rm(Arop, DQ, lane, k0);
      v16bf_t b;
      {
        int kb = k0 + ((lane >> 4) ? 16 : 0);
#pragma unroll
        for (int j = 0; j < 8; ++j) {
          int kk = kb + 2 * j;
          b[2 * j]     = (__bf16)krow[kk];
          b[2 * j + 1] = (__bf16)krow[kk + 1];
        }
      }
      acc = wmma_bf16(a, b, acc);
    }
    bool pad = (sidx[col] == T);
    int mOff = (lane >> 4) * 8;
#pragma unroll
    for (int i = 0; i < 8; ++i)
      sl[i + mOff][col] = pad ? NEGV : acc[i] * QSCALE;
  }
  __syncthreads();

  // ---- softmax per head ----
  for (int h = 0; h < H; ++h) {
    float m = -3.0e38f;
    for (int c = lane; c < TOPK; c += 32) m = fmaxf(m, sl[h][c]);
    for (int o = 16; o > 0; o >>= 1) m = fmaxf(m, __shfl_xor(m, o));
    float s = 0.f;
    for (int c = lane; c < TOPK; c += 32) {
      float e = __expf(sl[h][c] - m); sl[h][c] = e; s += e;
    }
    for (int o = 16; o > 0; o >>= 1) s += __shfl_xor(s, o);
    float inv = 1.0f / s;
    for (int c = lane; c < TOPK; c += 32) sp[h][c] = (__bf16)(sl[h][c] * inv);
  }
  __syncthreads();

  // ---- stage 2 ----
  for (int n0 = 0; n0 < LKV; n0 += 16) {
    v8f_t acc = {};
    int ncol = n0 + (lane & 15);
    for (int k0 = 0; k0 < TOPK; k0 += 32) {
      v16bf_t a;
      {
        int row = lane & 15;
        int kb = k0 + ((lane >> 4) ? 16 : 0);
#pragma unroll
        for (int j = 0; j < 8; ++j) {
          int kk = kb + 2 * j;
          a[2 * j] = sp[row][kk]; a[2 * j + 1] = sp[row][kk + 1];
        }
      }
      v16bf_t b;
      {
        int kb = k0 + ((lane >> 4) ? 16 : 0);
#pragma unroll
        for (int j = 0; j < 8; ++j) {
          int kk = kb + 2 * j;
          b[2 * j]     = (__bf16)skv[(long)sidx[kk] * KVW + ncol];
          b[2 * j + 1] = (__bf16)skv[(long)sidx[kk + 1] * KVW + ncol];
        }
      }
      acc = wmma_bf16(a, b, acc);
    }
    int mOff = (lane >> 4) * 8;
#pragma unroll
    for (int i = 0; i < 8; ++i)
      aout[((long)t * H + i + mOff) * LKV + ncol] = acc[i];
  }
}

// ---------------------------------------------------------------------------
extern "C" void kernel_launch(void* const* d_in, const int* in_sizes, int n_in,
                              void* d_out, int out_size, void* d_ws, size_t ws_size,
                              hipStream_t stream) {
  using namespace dsa;
  (void)in_sizes; (void)n_in; (void)out_size; (void)ws_size;

  const float* hidden = (const float*)d_in[0];   // [1,T,HID]
  const float* cosp   = (const float*)d_in[1];   // [T,DR]
  const float* sinp   = (const float*)d_in[2];   // [T,DR]
  const int*   ks     = (const int*)d_in[3];
  const int*   ke     = (const int*)d_in[4];
  const float* w_qa   = (const float*)d_in[5];   // [LQ,HID]
  const float* g_qa   = (const float*)d_in[6];
  const float* w_qb   = (const float*)d_in[7];   // [H*DQ,LQ]
  const float* w_kva  = (const float*)d_in[8];   // [LKV+DR,HID]
  const float* g_kva  = (const float*)d_in[9];
  const float* w_kvb  = (const float*)d_in[10];  // [H*(DN+DV),LKV]
  const float* w_o    = (const float*)d_in[11];  // [HID,H*DV]
  const float* i_wqb  = (const float*)d_in[12];  // [HI*DI,LQ]
  const float* i_wk   = (const float*)d_in[13];  // [DI,HID]
  const float* i_lnw  = (const float*)d_in[14];
  const float* i_lnb  = (const float*)d_in[15];
  const float* i_wpr  = (const float*)d_in[16];  // [HI,HID]

  // workspace layout (f32 elements)
  float* ws = (float*)d_ws;
  size_t off = 0;
  auto alloc = [&](size_t n) { float* p = ws + off; off += n; return p; };
  float* q_lat = alloc((size_t)T * LQ);
  float* skv   = alloc((size_t)(T + 1) * (LKV + DR));  // ckv / skv + pad row
  float* qi    = alloc((size_t)T * HI * DI);           // q_idx -> qi (roped)
  float* ki    = alloc((size_t)T * DI);                // k_idx -> ki
  float* wpj   = alloc((size_t)T * HI);
  float* score = alloc((size_t)T * T);
  int*   idx   = (int*)alloc((size_t)T * TOPK);
  float* qfull = alloc((size_t)T * H * DQ);
  float* qabs  = alloc((size_t)T * H * LKV);
  float* aout  = alloc((size_t)T * H * LKV);
  float* obuf  = alloc((size_t)T * H * DV);
  float* wkt   = alloc((size_t)H * LKV * DN);          // w_kn transposed
  float* outp  = (float*)d_out;                        // [T, HID]

  constexpr int KVW = LKV + DR;

  // 0) one-time weight transpose (L2-resident afterwards)
  dsa_transpose_wkn<<<dim3((LKV * DN) / 256, H), 256, 0, stream>>>(w_kvb, wkt);

  // 1) dense projections from hidden
  dsa_gemm_bf16<4><<<dim3(LQ / 64, T / 16, 1), 32, 0, stream>>>(
      hidden, HID, 0, w_qa, HID, 0, q_lat, LQ, 0, HID);
  dsa_gemm_bf16<4><<<dim3(KVW / 64, T / 16, 1), 32, 0, stream>>>(
      hidden, HID, 0, w_kva, HID, 0, skv, KVW, 0, HID);
  dsa_gemm_bf16<4><<<dim3(DI / 64, T / 16, 1), 32, 0, stream>>>(
      hidden, HID, 0, i_wk, HID, 0, ki, DI, 0, HID);
  dsa_gemm_bf16<1><<<dim3(HI / 16, T / 16, 1), 32, 0, stream>>>(
      hidden, HID, 0, i_wpr, HID, 0, wpj, HI, 0, HID);

  // 2) q_lat RMS, then q-side projections
  dsa_rms<<<T, 256, 0, stream>>>(q_lat, LQ, LQ, g_qa);
  dsa_gemm_bf16<4><<<dim3((HI * DI) / 64, T / 16, 1), 32, 0, stream>>>(
      q_lat, LQ, 0, i_wqb, LQ, 0, qi, HI * DI, 0, LQ);
  dsa_gemm_bf16<4><<<dim3((H * DQ) / 64, T / 16, 1), 32, 0, stream>>>(
      q_lat, LQ, 0, w_qb, LQ, 0, qfull, H * DQ, 0, LQ);

  // 3) norms, pad row, ropes
  dsa_rms<<<T, 256, 0, stream>>>(skv, KVW, LKV, g_kva);        // k_c
  dsa_ln<<<T, 128, 0, stream>>>(ki, DI, i_lnw, i_lnb);
  dsa_zero<<<(KVW + 255) / 256, 256, 0, stream>>>(skv + (size_t)T * KVW, KVW);
  dsa_rope<<<dim3(T, 1), 32, 0, stream>>>(skv + LKV, KVW, 0, cosp, sinp);  // k_r
  dsa_rope<<<dim3(T, HI), 32, 0, stream>>>(qi, HI * DI, DI, cosp, sinp);   // q_pe
  dsa_rope<<<dim3(T, 1), 32, 0, stream>>>(ki, DI, 0, cosp, sinp);          // k_pe
  dsa_rope<<<dim3(T, H), 32, 0, stream>>>(qfull + DN, H * DQ, DQ, cosp, sinp); // q_rope

  // 4) indexer scores + top-k
  dsa_indexer<<<dim3(T / 64, T), 32, 0, stream>>>(qi, ki, wpj, ks, ke, score);
  dsa_topk<<<T, 256, 0, stream>>>(score, idx);

  // 5) q_abs[t,h,:] = q_nope[t,h,:] @ wkt[h]^T   (per-head, B [N,K] rm)
  dsa_gemm_bf16<4><<<dim3(LKV / 64, T / 16, H), 32, 0, stream>>>(
      qfull, H * DQ, DQ,                 // A base advances by one head (DQ)
      wkt, DN, (long)LKV * DN,           // B = w_kn[h]^T, [LKV x DN]
      qabs, H * LKV, LKV, DN);

  // 6) sparse attention core
  dsa_attention<<<T, 32, 0, stream>>>(qabs, qfull, skv, idx, aout);

  // 7) o[t,h,:] = attn_out[t,h,:] @ w_v[h]^T    (per-head, B stored [N,K])
  dsa_gemm_bf16<4><<<dim3(DV / 64, T / 16, H), 32, 0, stream>>>(
      aout, H * LKV, LKV,
      w_kvb + (size_t)DN * LKV, LKV, (long)(DN + DV) * LKV,
      obuf, H * DV, DV, LKV);

  // 8) final projection
  dsa_gemm_bf16<4><<<dim3(HID / 64, T / 16, 1), 32, 0, stream>>>(
      obuf, H * DV, 0, w_o, H * DV, 0, outp, HID, 0, H * DV);
}
